// AttentionMemory_60653528154461
// MI455X (gfx1250) — matrix operand order repaired and checked by reading, not compiled, not run on hardware
//
#include <hip/hip_runtime.h>
#include <hip/hip_bf16.h>

// Problem dims (fixed by the reference)
#define B_   256
#define H_   1024
#define D_   256
#define G_   32
#define C_   16
#define O_   256
#define WROW (H_ + D_)   // attn_w row length = 1280 floats

typedef __attribute__((ext_vector_type(16))) __bf16 v16bf;
typedef __attribute__((ext_vector_type(8)))  __bf16 bf16x8;
typedef __attribute__((ext_vector_type(4)))  __bf16 bf16x4;
typedef __attribute__((ext_vector_type(8)))  float  v8f;
typedef __attribute__((ext_vector_type(4)))  float  f32x4;

// ---- fp32 -> bf16 (RNE) helpers for the hi/lo split ----
static __device__ __forceinline__ unsigned short bf16_rne_bits(float x) {
  unsigned u = __builtin_bit_cast(unsigned, x);
  unsigned r = u + 0x7FFFu + ((u >> 16) & 1u);
  return (unsigned short)(r >> 16);
}
static __device__ __forceinline__ __bf16 bits_to_bf16(unsigned short h) {
  return __builtin_bit_cast(__bf16, h);
}
static __device__ __forceinline__ float bf16bits_to_f32(unsigned short h) {
  return __builtin_bit_cast(float, (unsigned)h << 16);
}

#if defined(__has_builtin)
#if __has_builtin(__builtin_amdgcn_tanhf)
#define TANH_F32(x) __builtin_amdgcn_tanhf(x)
#endif
#endif
#ifndef TANH_F32
#define TANH_F32(x) tanhf(x)
#endif

// ------------------------------------------------------------------
// Kernel 1: split hidden (B,H) fp32 into bf16 hi/lo pair.
// ------------------------------------------------------------------
__global__ __launch_bounds__(256)
void prep_hidden_kernel(const float* __restrict__ hidden,
                        __bf16* __restrict__ hi, __bf16* __restrict__ lo) {
  int i = blockIdx.x * 256 + threadIdx.x;          // B_*H_ threads total
  float x = hidden[i];
  unsigned short hb = bf16_rne_bits(x);
  hi[i] = bits_to_bf16(hb);
  float l = x - bf16bits_to_f32(hb);
  lo[i] = bits_to_bf16(bf16_rne_bits(l));
}

// ------------------------------------------------------------------
// Kernel 2: stream-split attn_w[:, :, :H] into dense bf16 hi/lo arrays
// with row stride H (enc columns stripped). Pure bandwidth pass:
// 64 MB fp32 in -> 2 x 32 MB bf16 out; conversion VALU hides under HBM.
// One thread per 4 elements (float4 in, bf16x4 out x2).
// ------------------------------------------------------------------
__global__ __launch_bounds__(256)
void prep_w_kernel(const float* __restrict__ attn_w,
                   __bf16* __restrict__ whi, __bf16* __restrict__ wlo) {
  size_t t   = (size_t)blockIdx.x * 256 + threadIdx.x; // < C*H*H/4
  size_t row = t >> 8;                                 // 0 .. C*H-1
  int    k4  = (int)(t & 255) << 2;                    // 0,4,...,1020
  f32x4 x = *(const f32x4*)(attn_w + row * WROW + k4);
  bf16x4 h4, l4;
  #pragma unroll
  for (int i = 0; i < 4; ++i) {
    unsigned short hb = bf16_rne_bits(x[i]);
    h4[i] = bits_to_bf16(hb);
    l4[i] = bits_to_bf16(bf16_rne_bits(x[i] - bf16bits_to_f32(hb)));
  }
  *(bf16x4*)(whi + row * H_ + k4) = h4;
  *(bf16x4*)(wlo + row * H_ + k4) = l4;
}

// ------------------------------------------------------------------
// Kernel 3: enc_part[c,g,o] = sum_d gst[c,g,d] * attn_w[c,o,H+d]
// 268 MFLOP total -> plain VALU; gst reads are wave-uniform.
// ------------------------------------------------------------------
__global__ __launch_bounds__(256)
void enc_part_kernel(const float* __restrict__ gst,
                     const float* __restrict__ attn_w,
                     float* __restrict__ ep) {
  const int c = blockIdx.y;
  const int o = blockIdx.x * 256 + threadIdx.x;    // 0..H_-1
  const float* wrow = attn_w + ((size_t)c * H_ + o) * WROW + H_;
  const float* gc   = gst + (size_t)c * G_ * D_;
  float acc[G_];
  #pragma unroll
  for (int g = 0; g < G_; ++g) acc[g] = 0.f;
  for (int d = 0; d < D_; ++d) {
    float wv = wrow[d];
    #pragma unroll
    for (int g = 0; g < G_; ++g) acc[g] += gc[g * D_ + d] * wv;
  }
  for (int g = 0; g < G_; ++g)
    ep[((size_t)c * G_ + g) * H_ + o] = acc[g];
}

// ------------------------------------------------------------------
// Kernel 4: hid_part[c,b,o] = hidden(b,:) . w_hid[c,o,:] + attn_b[c,o]
// All-bf16-operand WMMA GEMM (bf16x3 fp32 emulation), no conversion
// VALU in the hot loop. Block = 256 threads (8 waves); each wave owns a
// 16(b) x 64(o) tile (4 accumulators sharing one A operand).
// K loop: 32 steps of 32 -> 12 v_wmma per step vs ~20 b128 loads.
// ------------------------------------------------------------------
__global__ __launch_bounds__(256)
void gemm_hid_kernel(const __bf16* __restrict__ whi,
                     const __bf16* __restrict__ wlo,
                     const float* __restrict__ attn_b,
                     const __bf16* __restrict__ hhi,
                     const __bf16* __restrict__ hlo,
                     float* __restrict__ hp) {
  const int c     = blockIdx.z;
  const int b0    = blockIdx.y << 4;                 // 16 rows of hidden
  const int wave  = threadIdx.x >> 5;
  const int lane  = threadIdx.x & 31;
  const int nlane = lane & 15;
  const int half  = lane >> 4;
  const int o0    = (blockIdx.x << 9) + (wave << 6); // 64 o-cols per wave

  // A operand rows (16-bit A 16x32, contiguous K order used here: both lane halves carry M = lane%16,
  // lane<16 holds K {0..15}, lane>=16 holds K {16..31}).
  const __bf16* arow_hi = hhi + (size_t)(b0 + nlane) * H_;
  const __bf16* arow_lo = hlo + (size_t)(b0 + nlane) * H_;
  // B operand (32x16): lane n = lane%16 is w-row (o0 + t*16 + n);
  // lanes 0-15 carry K k0..k0+15, lanes 16-31 carry K k0+16..k0+31.
  const __bf16* brow_hi = whi + ((size_t)c * H_ + (o0 + nlane)) * H_;
  const __bf16* brow_lo = wlo + ((size_t)c * H_ + (o0 + nlane)) * H_;

  v8f acc[4] = {};

  for (int k0 = 0; k0 < H_; k0 += 32) {
    const int ka = k0 + (half << 4);
    bf16x8 a0h = *(const bf16x8*)(arow_hi + ka);
    bf16x8 a1h = *(const bf16x8*)(arow_hi + ka + 8);
    bf16x8 a0l = *(const bf16x8*)(arow_lo + ka);
    bf16x8 a1l = *(const bf16x8*)(arow_lo + ka + 8);
    v16bf Ahi = __builtin_shufflevector(a0h, a1h, 0,1,2,3,4,5,6,7,8,9,10,11,12,13,14,15);
    v16bf Alo = __builtin_shufflevector(a0l, a1l, 0,1,2,3,4,5,6,7,8,9,10,11,12,13,14,15);

    const int kb = k0 + (half << 4);
    #pragma unroll
    for (int t = 0; t < 4; ++t) {
      const __bf16* ph = brow_hi + (size_t)(t << 4) * H_ + kb;
      const __bf16* pl = brow_lo + (size_t)(t << 4) * H_ + kb;
      v16bf Bhi = *(const v16bf*)ph;   // 32B contiguous per lane
      v16bf Blo = *(const v16bf*)pl;
      __builtin_prefetch((const void*)(ph + 128), 0, 0);  // K tile 4 steps ahead -> global_prefetch_b8
      // fp32-accuracy product: Ahi*Bhi + Ahi*Blo + Alo*Bhi (Alo*Blo ~ 2^-16, dropped)
      acc[t] = __builtin_amdgcn_wmma_f32_16x16x32_bf16(false, Ahi, false, Bhi, (short)0, acc[t], false, false);
      acc[t] = __builtin_amdgcn_wmma_f32_16x16x32_bf16(false, Ahi, false, Blo, (short)0, acc[t], false, false);
      acc[t] = __builtin_amdgcn_wmma_f32_16x16x32_bf16(false, Alo, false, Bhi, (short)0, acc[t], false, false);
    }
  }

  // Epilogue: D layout -> lane L: col = L%16, VGPR r: row = r + 8*(L>=16).
  #pragma unroll
  for (int t = 0; t < 4; ++t) {
    const int col  = o0 + (t << 4) + nlane;
    const float bias = attn_b[c * H_ + col];
    #pragma unroll
    for (int r = 0; r < 8; ++r) {
      const int row = b0 + r + (half << 3);
      hp[((size_t)c * B_ + row) * H_ + col] = acc[t][r] + bias;
    }
  }
}

// ------------------------------------------------------------------
// Kernel 5: scores[b,c,g] = sum_h tanh(hp[c,b,h] + ep[c,g,h]) * v[c,h],
// then softmax over g. One block per (b,c); 8 waves x 4 g each.
// The 512 MB energy tensor is never materialized.
// ------------------------------------------------------------------
__global__ __launch_bounds__(256)
void scores_kernel(const float* __restrict__ hp,
                   const float* __restrict__ ep,
                   const float* __restrict__ v,
                   float* __restrict__ att) {
  const int b = blockIdx.x, c = blockIdx.y;
  const int wave = threadIdx.x >> 5, lane = threadIdx.x & 31;
  __shared__ float s_sc[G_];

  const float* hprow = hp + ((size_t)c * B_ + b) * H_;
  const float* vrow  = v + (size_t)c * H_;
  const float* ep0   = ep + ((size_t)c * G_ + wave * 4) * H_;

  float a0 = 0.f, a1 = 0.f, a2 = 0.f, a3 = 0.f;
  for (int i = lane; i < H_; i += 32) {
    float hv = hprow[i];
    float vv = vrow[i];
    a0 += TANH_F32(hv + ep0[i])           * vv;
    a1 += TANH_F32(hv + ep0[H_ + i])      * vv;
    a2 += TANH_F32(hv + ep0[2 * H_ + i])  * vv;
    a3 += TANH_F32(hv + ep0[3 * H_ + i])  * vv;
  }
  #pragma unroll
  for (int off = 16; off > 0; off >>= 1) {
    a0 += __shfl_xor(a0, off, 32);
    a1 += __shfl_xor(a1, off, 32);
    a2 += __shfl_xor(a2, off, 32);
    a3 += __shfl_xor(a3, off, 32);
  }
  if (lane == 0) {
    s_sc[wave * 4 + 0] = a0;
    s_sc[wave * 4 + 1] = a1;
    s_sc[wave * 4 + 2] = a2;
    s_sc[wave * 4 + 3] = a3;
  }
  __syncthreads();
  if (threadIdx.x < G_) {
    float s = s_sc[threadIdx.x];
    float m = s;
    #pragma unroll
    for (int off = 16; off > 0; off >>= 1) m = fmaxf(m, __shfl_xor(m, off, 32));
    float e = expf(s - m);
    float tot = e;
    #pragma unroll
    for (int off = 16; off > 0; off >>= 1) tot += __shfl_xor(tot, off, 32);
    att[(size_t)b * (C_ * G_) + c * G_ + threadIdx.x] = e / tot;
  }
}

// ------------------------------------------------------------------
// Kernel 6: weighted/pool/output. One block per b.
// pooled[d] = (1/C) sum_{c,g} att[b,c,g]*gst[c,g,d]; out = pooled@out_w^T + out_b
// ------------------------------------------------------------------
__global__ __launch_bounds__(256)
void out_kernel(const float* __restrict__ att,
                const float* __restrict__ gst,
                const float* __restrict__ out_w,
                const float* __restrict__ out_b,
                float* __restrict__ out) {
  const int b = blockIdx.x, tid = threadIdx.x;
  __shared__ float s_att[C_ * G_];
  __shared__ float s_pool[D_];
  s_att[tid]       = att[(size_t)b * (C_ * G_) + tid];
  s_att[tid + 256] = att[(size_t)b * (C_ * G_) + tid + 256];
  __syncthreads();

  float p = 0.f;
  for (int cg = 0; cg < C_ * G_; ++cg)
    p += s_att[cg] * gst[(size_t)cg * D_ + tid];      // tid == d, coalesced
  s_pool[tid] = p * (1.0f / C_);
  __syncthreads();

  float acc = out_b[tid];                              // tid == o
  const float* wr = out_w + (size_t)tid * D_;
  for (int d = 0; d < D_; ++d) acc += s_pool[d] * wr[d];
  out[(size_t)b * O_ + tid] = acc;
}

// ------------------------------------------------------------------
extern "C" void kernel_launch(void* const* d_in, const int* in_sizes, int n_in,
                              void* d_out, int out_size, void* d_ws, size_t ws_size,
                              hipStream_t stream) {
  (void)in_sizes; (void)n_in; (void)out_size; (void)ws_size;
  const float* hidden = (const float*)d_in[0];
  const float* gst    = (const float*)d_in[1];
  const float* attn_w = (const float*)d_in[2];
  const float* attn_b = (const float*)d_in[3];
  const float* v      = (const float*)d_in[4];
  const float* out_w  = (const float*)d_in[5];
  const float* out_b  = (const float*)d_in[6];
  float* out = (float*)d_out;

  // Workspace layout (~83 MB total):
  //  w_hi   bf16 C*H*H   (32 MB)
  //  w_lo   bf16 C*H*H   (32 MB)
  //  hid_hi bf16 B*H     (512 KB)
  //  hid_lo bf16 B*H     (512 KB)
  //  enc    f32  C*G*H   (2 MB)
  //  hid_pt f32  C*B*H   (16 MB)
  char* ws = (char*)d_ws;
  const size_t wsz = (size_t)C_ * H_ * H_ * 2;   // 32 MB per w array
  __bf16* whi = (__bf16*)ws;
  __bf16* wlo = (__bf16*)(ws + wsz);
  __bf16* hhi = (__bf16*)(ws + 2 * wsz);
  __bf16* hlo = (__bf16*)(ws + 2 * wsz + (size_t)B_ * H_ * 2);
  float*  ep  = (float*)(ws + 2 * wsz + (size_t)B_ * H_ * 4);
  float*  hp  = (float*)(ws + 2 * wsz + (size_t)B_ * H_ * 4 + (size_t)C_ * G_ * H_ * 4);

  prep_hidden_kernel<<<(B_ * H_) / 256, 256, 0, stream>>>(hidden, hhi, hlo);
  prep_w_kernel<<<((size_t)C_ * H_ * H_ / 4) / 256, 256, 0, stream>>>(attn_w, whi, wlo);
  enc_part_kernel<<<dim3(H_ / 256, C_), 256, 0, stream>>>(gst, attn_w, ep);
  gemm_hid_kernel<<<dim3(H_ / 512, B_ / 16, C_), 256, 0, stream>>>(whi, wlo, attn_b, hhi, hlo, hp);
  scores_kernel<<<dim3(B_, C_), 256, 0, stream>>>(hp, ep, v, out);
  out_kernel<<<B_, 256, 0, stream>>>(out, gst, out_w, out_b, out + (size_t)B_ * C_ * G_);
}
